// SparseSelfAttention_45835890983061
// MI455X (gfx1250) — compile-verified, matrix operand order checked by reading, not run on hardware
//
#include <hip/hip_runtime.h>
#include <hip/hip_bf16.h>
#include <cstdint>

// ---------------------------------------------------------------------------
// Problem constants (from the reference)
// ---------------------------------------------------------------------------
#define HEADS  16
#define SPH    64
#define KEYD   64
#define EMB    1024
#define BATCH  2
#define LSEQ   4096
#define NEWLEN 4098          // LSEQ padded to multiple of STEP=3
#define NBLK   1366          // NEWLEN / 3
#define NBLKP  1376          // NBLK padded to multiple of 32 (key tiles)
#define NTILES (NBLKP / 32)  // 43 key tiles
#define NROWS  (BATCH * NEWLEN)   // 8196
#define NROWSP 8224               // padded to multiple of 32 (GEMM row tiles)
#define QSCALE 0.125f             // 1/sqrt(KEYD), folded into W_q
#define NEG_BIG (-3.0e38f)

typedef __attribute__((ext_vector_type(16))) __bf16 v16bf;
typedef __attribute__((ext_vector_type(8)))  float  v8f;

union Frag16 {
    v16bf v;
    uint4 u[2];
};
union BF8 {
    uint4  u;
    __bf16 h[8];
};

// Load a 16x32 bf16 WMMA fragment (A layout; B 32x16 is the same pattern with
// "row" = output column). Element (i, k) comes from base[i*ld + k].
// Lanes 0-15 hold i=lane, k in {0..7} u {16..23}; lanes 16-31 hold the
// {8..15} u {24..31} sub-chunks (CDNA5 ISA 7.12.2). Two b128 loads per lane.
__device__ __forceinline__ Frag16 load_frag16x32(const __bf16* base, int ld) {
    const int lane = threadIdx.x & 31;
    const int i    = lane & 15;
    const int hi   = lane >> 4;
    const __bf16* p = base + (size_t)i * ld + hi * 8;
    Frag16 f;
    f.u[0] = *reinterpret_cast<const uint4*>(p);
    f.u[1] = *reinterpret_cast<const uint4*>(p + 16);
    return f;
}

__device__ __forceinline__ v8f wmma_bf16(const Frag16& a, const Frag16& b, v8f c) {
    return __builtin_amdgcn_wmma_f32_16x16x32_bf16(
        /*neg_a=*/false, a.v, /*neg_b=*/false, b.v,
        /*c_mod=*/(short)0, c, /*reuse_a=*/false, /*reuse_b=*/false);
}

__device__ __forceinline__ v8f v8f_zero() {
    v8f z;
#pragma unroll
    for (int r = 0; r < 8; ++r) z[r] = 0.0f;
    return z;
}

// Butterfly reductions across the 16-lane half that owns one C/D-matrix row.
__device__ __forceinline__ float half16_max(float t) {
    t = fmaxf(t, __shfl_xor(t, 1, 32));
    t = fmaxf(t, __shfl_xor(t, 2, 32));
    t = fmaxf(t, __shfl_xor(t, 4, 32));
    t = fmaxf(t, __shfl_xor(t, 8, 32));
    return t;
}
__device__ __forceinline__ float half16_sum(float t) {
    t += __shfl_xor(t, 1, 32);
    t += __shfl_xor(t, 2, 32);
    t += __shfl_xor(t, 4, 32);
    t += __shfl_xor(t, 8, 32);
    return t;
}

// ---------------------------------------------------------------------------
// 1) fp32 -> bf16 conversion of x (zero padded to NROWSP rows)
// ---------------------------------------------------------------------------
__global__ void conv_x_kernel(const float* __restrict__ x, __bf16* __restrict__ xb) {
    int idx = blockIdx.x * blockDim.x + threadIdx.x;
    if (idx >= NROWSP * EMB) return;
    int row = idx / EMB;
    int k   = idx - row * EMB;
    float v = 0.0f;
    if (row < NROWS) {
        int b   = row / NEWLEN;
        int pos = row - b * NEWLEN;
        if (pos < LSEQ) v = x[((size_t)(b * LSEQ + pos)) * EMB + k];
    }
    xb[idx] = (__bf16)v;
}

// W (k-major [EMB][EMB]) -> bf16 transposed WT[n][k] so B fragments are
// contiguous b128 loads. scale folds 1/sqrt(KEYD) into W_q.
__global__ void conv_w_kernel(const float* __restrict__ w, __bf16* __restrict__ wt,
                              float scale) {
    int idx = blockIdx.x * blockDim.x + threadIdx.x;
    if (idx >= EMB * EMB) return;
    int n = idx / EMB;
    int k = idx - n * EMB;
    wt[idx] = (__bf16)(w[(size_t)k * EMB + n] * scale);
}

// ---------------------------------------------------------------------------
// 2) QKV projection GEMM: Y = xb @ W  (bf16 in, fp32 acc, bf16 out).
//    One wave computes a 32x64 output tile (2 A-frags x 4 B-frags -> 8 acc),
//    with register double-buffering of the next K-chunk so global_load_b128s
//    stay in flight across the 8 WMMAs of the current chunk.
//    Outputs scattered into phase-separated layouts:
//      Q,K : [b][s][h][NBLKP][64]       (row-major feature)
//      V   : [b][s][h][64][NBLKP]       (transposed: contiguous along blocks)
// ---------------------------------------------------------------------------
__global__ __launch_bounds__(128) void qkv_gemm_kernel(
    const __bf16* __restrict__ xb,
    const __bf16* __restrict__ wtq, const __bf16* __restrict__ wtk,
    const __bf16* __restrict__ wtv,
    __bf16* __restrict__ qp, __bf16* __restrict__ kp, __bf16* __restrict__ vt) {
    const int w    = threadIdx.x >> 5;
    const int lane = threadIdx.x & 31;
    const int i16  = lane & 15;
    const int hi   = lane >> 4;
    const int rt   = blockIdx.x;                 // 32-row tile
    const int cb   = (blockIdx.y * 4 + w) * 64;  // column base of this wave
    const int mat  = blockIdx.z;                 // 0=Q 1=K 2=V

    const __bf16* W = (mat == 0) ? wtq : (mat == 1) ? wtk : wtv;
    const __bf16* xr0 = xb + (size_t)rt * 32 * EMB;
    const __bf16* xr1 = xr0 + 16 * EMB;

    v8f acc[2][4];
#pragma unroll
    for (int rg = 0; rg < 2; ++rg)
#pragma unroll
        for (int nt = 0; nt < 4; ++nt) acc[rg][nt] = v8f_zero();

    // prologue loads (chunk 0)
    Frag16 a0 = load_frag16x32(xr0, EMB);
    Frag16 a1 = load_frag16x32(xr1, EMB);
    Frag16 bf[4];
#pragma unroll
    for (int nt = 0; nt < 4; ++nt)
        bf[nt] = load_frag16x32(W + (size_t)(cb + nt * 16) * EMB, EMB);

    for (int c = 0; c < EMB; c += 32) {
        const int cn = (c + 32) & (EMB - 1);   // wraps to 0 on last iter (harmless)
        // issue next chunk's loads before consuming the current fragments
        Frag16 a0n = load_frag16x32(xr0 + cn, EMB);
        Frag16 a1n = load_frag16x32(xr1 + cn, EMB);
        Frag16 bfn[4];
#pragma unroll
        for (int nt = 0; nt < 4; ++nt)
            bfn[nt] = load_frag16x32(W + (size_t)(cb + nt * 16) * EMB + cn, EMB);

#pragma unroll
        for (int nt = 0; nt < 4; ++nt) {
            acc[0][nt] = wmma_bf16(a0, bf[nt], acc[0][nt]);
            acc[1][nt] = wmma_bf16(a1, bf[nt], acc[1][nt]);
        }
        a0 = a0n;
        a1 = a1n;
#pragma unroll
        for (int nt = 0; nt < 4; ++nt) bf[nt] = bfn[nt];
    }

#pragma unroll
    for (int rg = 0; rg < 2; ++rg) {
#pragma unroll
        for (int r = 0; r < 8; ++r) {
            int row = rt * 32 + rg * 16 + r + 8 * hi;
            if (row >= NROWS) continue;
            int b   = row / NEWLEN;
            int rem = row - b * NEWLEN;
            int blk = rem / 3;
            int s   = rem - blk * 3;
#pragma unroll
            for (int nt = 0; nt < 4; ++nt) {
                int col = cb + nt * 16 + i16;
                int h   = col >> 6;
                int f   = col & 63;
                size_t plane = (size_t)((b * 3 + s) * HEADS + h);
                __bf16 val = (__bf16)acc[rg][nt][r];
                if (mat == 0)      qp[plane * (NBLKP * KEYD) + (size_t)blk * KEYD + f] = val;
                else if (mat == 1) kp[plane * (NBLKP * KEYD) + (size_t)blk * KEYD + f] = val;
                else               vt[plane * (SPH * NBLKP) + (size_t)f * NBLKP + blk] = val;
            }
        }
    }
}

// ---------------------------------------------------------------------------
// 3) Flash attention per (b,h,s): 1366 queries x (1366 global + 5 local) keys.
//    One wave = 16 queries; 2 waves/block; loop over 43 key tiles of 32.
//    V-fragments are loaded at loop top (consumed only after the barrier) and
//    the next tile's K-fragments are prefetched during the softmax section.
// ---------------------------------------------------------------------------
__global__ __launch_bounds__(64) void attn_kernel(
    const __bf16* __restrict__ Qp, const __bf16* __restrict__ Kp,
    const __bf16* __restrict__ Vt, float* __restrict__ out) {
    __shared__ __align__(16) __bf16 pbuf[2][16][32];  // P tile (softmax probs)
    __shared__ float tloc[2][16][5];                  // local neighbor scores

    const int w    = threadIdx.x >> 5;
    const int lane = threadIdx.x & 31;
    const int i16  = lane & 15;
    const int hi   = lane >> 4;

    const int bhs = blockIdx.y;          // = ((b*3+s)*16+h), 0..95
    const int b   = bhs / 48;
    const int s   = (bhs / 16) % 3;
    const int h   = bhs & 15;
    const int qtile = blockIdx.x * 2 + w;  // 0..85
    const int qb0   = qtile * 16;

    const __bf16* Qb = Qp + (size_t)bhs * NBLKP * KEYD;
    const __bf16* Kb = Kp + (size_t)bhs * NBLKP * KEYD;
    const __bf16* Vb = Vt + (size_t)bhs * SPH * NBLKP;

    // Q fragments for this wave's 16 query rows (K-dim 64 = 2 chunks)
    Frag16 aq0 = load_frag16x32(Qb + (size_t)qb0 * KEYD, KEYD);
    Frag16 aq1 = load_frag16x32(Qb + (size_t)qb0 * KEYD + 32, KEYD);

    float m_r[8], l_r[8];
    v8f acc[4];
#pragma unroll
    for (int r = 0; r < 8; ++r) { m_r[r] = NEG_BIG; l_r[r] = 0.0f; }
#pragma unroll
    for (int d = 0; d < 4; ++d) acc[d] = v8f_zero();

    // prologue: K fragments of tile 0
    Frag16 bk[4];
    bk[0] = load_frag16x32(Kb, KEYD);
    bk[1] = load_frag16x32(Kb + 32, KEYD);
    bk[2] = load_frag16x32(Kb + 16 * KEYD, KEYD);
    bk[3] = load_frag16x32(Kb + 16 * KEYD + 32, KEYD);

    for (int kt = 0; kt < NTILES; ++kt) {
        const int kb  = kt * 32;
        const int kbn = (kt + 1 < NTILES) ? kb + 32 : kb;

        // V fragments for the current tile: consumed only after the barrier,
        // so these loads overlap the score WMMAs and the softmax section.
        Frag16 bv[4];
#pragma unroll
        for (int d = 0; d < 4; ++d)
            bv[d] = load_frag16x32(Vb + (size_t)(d * 16) * NBLKP + kb, NBLKP);

        // S[16q x 32k] = Q @ K^T
        v8f s0 = wmma_bf16(aq0, bk[0], v8f_zero());
        s0 = wmma_bf16(aq1, bk[1], s0);
        v8f s1 = wmma_bf16(aq0, bk[2], v8f_zero());
        s1 = wmma_bf16(aq1, bk[3], s1);

        // prefetch next tile's K fragments (in flight during softmax)
        Frag16 bkn[4];
        bkn[0] = load_frag16x32(Kb + (size_t)kbn * KEYD, KEYD);
        bkn[1] = load_frag16x32(Kb + (size_t)kbn * KEYD + 32, KEYD);
        bkn[2] = load_frag16x32(Kb + (size_t)(kbn + 16) * KEYD, KEYD);
        bkn[3] = load_frag16x32(Kb + (size_t)(kbn + 16) * KEYD + 32, KEYD);

        // mask padded keys (>= NBLK) so they do not enter the softmax
        if (kb + i16 >= NBLK) {
#pragma unroll
            for (int r = 0; r < 8; ++r) s0[r] = NEG_BIG;
        }
        if (kb + 16 + i16 >= NBLK) {
#pragma unroll
            for (int r = 0; r < 8; ++r) s1[r] = NEG_BIG;
        }

        // online softmax update per row (row = r + 8*hi, replicated on 16 lanes)
#pragma unroll
        for (int r = 0; r < 8; ++r) {
            float rm = half16_max(fmaxf(s0[r], s1[r]));
            float mn = fmaxf(m_r[r], rm);
            float sc = __expf(m_r[r] - mn);
            float p0 = __expf(s0[r] - mn);
            float p1 = __expf(s1[r] - mn);
            float rs = half16_sum(p0 + p1);
            l_r[r] = l_r[r] * sc + rs;
            m_r[r] = mn;
#pragma unroll
            for (int d = 0; d < 4; ++d) acc[d][r] *= sc;
            int row = r + 8 * hi;
            pbuf[w][row][i16]      = (__bf16)p0;
            pbuf[w][row][16 + i16] = (__bf16)p1;
        }
        __syncthreads();   // D-layout -> A-layout transpose goes through LDS

        Frag16 ap = load_frag16x32(&pbuf[w][0][0], 32);
#pragma unroll
        for (int d = 0; d < 4; ++d)
            acc[d] = wmma_bf16(ap, bv[d], acc[d]);
        __syncthreads();   // pbuf reused next iteration

#pragma unroll
        for (int d = 0; d < 4; ++d) bk[d] = bkn[d];
    }

    // ---- 5 local neighbor keys (same phase, blocks q-2..q+2). In-range
    // score = q.k (identical dot product as global); out-of-range score = 0
    // (reference zero-pads k), and its value contribution is 0.
    if (lane < 16) {
        const uint4* q4 = reinterpret_cast<const uint4*>(Qb + (size_t)(qb0 + i16) * KEYD);
        BF8 qv[8];
#pragma unroll
        for (int j = 0; j < 8; ++j) qv[j].u = q4[j];
#pragma unroll
        for (int d = 0; d < 5; ++d) {
            int nb = qb0 + i16 + d - 2;
            float dot = 0.0f;
            if (nb >= 0 && nb < NBLK) {
                const uint4* k4 = reinterpret_cast<const uint4*>(Kb + (size_t)nb * KEYD);
#pragma unroll
                for (int j = 0; j < 8; ++j) {
                    BF8 kv; kv.u = k4[j];
#pragma unroll
                    for (int e = 0; e < 8; ++e)
                        dot += (float)qv[j].h[e] * (float)kv.h[e];
                }
            }
            tloc[w][i16][d] = dot;
        }
    }
    __syncthreads();

#pragma unroll
    for (int r = 0; r < 8; ++r) {
        int row = r + 8 * hi;
        float tv[5], tmax = NEG_BIG;
#pragma unroll
        for (int d = 0; d < 5; ++d) { tv[d] = tloc[w][row][d]; tmax = fmaxf(tmax, tv[d]); }
        float mn = fmaxf(m_r[r], tmax);
        float sc = __expf(m_r[r] - mn);
        float e[5], ls = 0.0f;
#pragma unroll
        for (int d = 0; d < 5; ++d) { e[d] = __expf(tv[d] - mn); ls += e[d]; }
        l_r[r] = l_r[r] * sc + ls;
        m_r[r] = mn;
#pragma unroll
        for (int dt = 0; dt < 4; ++dt) acc[dt][r] *= sc;
#pragma unroll
        for (int d = 0; d < 5; ++d) {
            int nb = qb0 + row + d - 2;
            if (nb >= 0 && nb < NBLK) {
#pragma unroll
                for (int dt = 0; dt < 4; ++dt)
                    acc[dt][r] += e[d] * (float)Vb[(size_t)(dt * 16 + i16) * NBLKP + nb];
            }
        }
    }

    // normalize and store (only the first LSEQ positions survive the crop)
#pragma unroll
    for (int r = 0; r < 8; ++r) {
        int row  = r + 8 * hi;
        int qblk = qb0 + row;
        int pos  = qblk * 3 + s;
        if (pos < LSEQ) {
            float inv = 1.0f / l_r[r];
            size_t ob = ((size_t)(b * LSEQ + pos)) * (HEADS * SPH) + h * SPH;
#pragma unroll
            for (int d = 0; d < 4; ++d)
                out[ob + d * 16 + i16] = acc[d][r] * inv;
        }
    }
}

// ---------------------------------------------------------------------------
// Host-side launch
// ---------------------------------------------------------------------------
extern "C" void kernel_launch(void* const* d_in, const int* in_sizes, int n_in,
                              void* d_out, int out_size, void* d_ws, size_t ws_size,
                              hipStream_t stream) {
    (void)in_sizes; (void)n_in; (void)out_size; (void)ws_size;
    const float* x  = (const float*)d_in[0];
    const float* wq = (const float*)d_in[1];
    const float* wk = (const float*)d_in[2];
    const float* wv = (const float*)d_in[3];
    float* out = (float*)d_out;

    char* ws = (char*)d_ws;
    size_t off = 0;
    auto carve = [&](size_t bytes) {
        char* p = ws + off;
        off += (bytes + 255) & ~(size_t)255;
        return p;
    };
    const size_t XB_BYTES    = (size_t)NROWSP * EMB * 2;
    const size_t WT_BYTES    = (size_t)EMB * EMB * 2;
    const size_t PHASE_BYTES = (size_t)BATCH * 3 * HEADS * NBLKP * 64 * 2;

    __bf16* xb  = (__bf16*)carve(XB_BYTES);
    __bf16* wtq = (__bf16*)carve(WT_BYTES);
    __bf16* wtk = (__bf16*)carve(WT_BYTES);
    __bf16* wtv = (__bf16*)carve(WT_BYTES);
    __bf16* qp  = (__bf16*)carve(PHASE_BYTES);
    __bf16* kp  = (__bf16*)carve(PHASE_BYTES);
    __bf16* vt  = (__bf16*)carve(PHASE_BYTES);

    // zero the phase buffers (pad blocks 1366..1375 are never written by GEMM)
    hipMemsetAsync(qp, 0, PHASE_BYTES, stream);
    hipMemsetAsync(kp, 0, PHASE_BYTES, stream);
    hipMemsetAsync(vt, 0, PHASE_BYTES, stream);

    conv_x_kernel<<<(NROWSP * EMB + 255) / 256, 256, 0, stream>>>(x, xb);
    conv_w_kernel<<<(EMB * EMB + 255) / 256, 256, 0, stream>>>(wq, wtq, QSCALE);
    conv_w_kernel<<<(EMB * EMB + 255) / 256, 256, 0, stream>>>(wk, wtk, 1.0f);
    conv_w_kernel<<<(EMB * EMB + 255) / 256, 256, 0, stream>>>(wv, wtv, 1.0f);

    qkv_gemm_kernel<<<dim3(NROWSP / 32, 4, 3), 128, 0, stream>>>(
        xb, wtq, wtk, wtv, qp, kp, vt);

    attn_kernel<<<dim3(43, BATCH * 3 * HEADS), 64, 0, stream>>>(qp, kp, vt, out);
}
